// GCL_Encoder_33663953666677
// MI455X (gfx1250) — matrix operand order repaired and checked by reading, not compiled, not run on hardware
//
#include <hip/hip_runtime.h>
#include <hip/hip_bf16.h>

#define IN_F  165
#define HID_F 128
#define OUT_F 64
#define EPS_BN 1e-5f

typedef __attribute__((ext_vector_type(16))) _Float16 v16h;
typedef __attribute__((ext_vector_type(8)))  float    v8f;

// ---------------------------------------------------------------------------
// Utility fills
// ---------------------------------------------------------------------------
__global__ void k_fill(float* __restrict__ p, long long n, float v) {
  long long i = (long long)blockIdx.x * blockDim.x + threadIdx.x;
  if (i < n) p[i] = v;
}

// deg[dst] += 1 per edge (deg pre-filled with 1.0 for self-loop)
__global__ void k_deg(const int* __restrict__ dst, int e, float* __restrict__ deg) {
  int i = blockIdx.x * blockDim.x + threadIdx.x;
  if (i < e) atomicAdd(&deg[dst[i]], 1.0f);
}

// deg -> dinv in place
__global__ void k_dinv(float* __restrict__ deg, int n) {
  int i = blockIdx.x * blockDim.x + threadIdx.x;
  if (i < n) {
    float d = deg[i];
    deg[i] = (d > 0.0f) ? rsqrtf(d) : 0.0f;
  }
}

// ---------------------------------------------------------------------------
// Pack W[K x F] (f32) into WMMA B-fragment order, f16 hi/lo split, K padded
// to a multiple of 32 with zeros. Fragment group g = (kc*2 + half)*F + col
// holds 16 consecutive halves (K = kc*32 + (i>>3)*16 + half*8 + (i&7)), so each GEMM lane
// reads its whole v16h fragment as one contiguous 32B vector load.
// ---------------------------------------------------------------------------
__global__ void k_pack_w(const float* __restrict__ W, int K, int F, int Kpad,
                         _Float16* __restrict__ Bh, _Float16* __restrict__ Bl) {
  int idx = blockIdx.x * blockDim.x + threadIdx.x;   // over Kpad*F
  if (idx >= Kpad * F) return;
  int k = idx / F;
  int col = idx - k * F;
  int kc = k >> 5, r = k & 31, h = (r >> 3) & 1, i = (r & 7) + ((r >> 4) << 3);
  float v = (k < K) ? W[(size_t)k * F + col] : 0.0f;
  _Float16 vh = (_Float16)v;
  size_t o = (((size_t)(kc * 2 + h) * F + col) << 4) + i;
  Bh[o] = vh;
  Bl[o] = (_Float16)(v - (float)vh);
}

// ---------------------------------------------------------------------------
// WMMA GEMM: C[n x F] = A[n x K] @ W[K x F] (+ bias, + optional ReLU)
// One wave computes a 16x64 tile (4 accumulators) so the A hi/lo fragments
// are reused 4x. f32 A is split hi/lo f16 in-kernel; all loads in main loop
// AND tail are branch-free (row clamp + index clamp + VALU zero-select).
// B comes pre-packed as f16 hi/lo fragments -> pure b128 vector loads.
// 12 x v_wmma_f32_16x16x32_f16 per K-chunk; f32-class precision via the
// 3-product Markidis split. Wave-uniform exit keeps EXEC all-1s at WMMAs.
// ---------------------------------------------------------------------------
__global__ void k_wmma_gemm(const float* __restrict__ A,
                            const _Float16* __restrict__ Bh,
                            const _Float16* __restrict__ Bl,
                            const float* __restrict__ bias,
                            float* __restrict__ C,
                            int n, int K, int F, int fuse_relu) {
  const int lane = threadIdx.x & 31;
  const int wave = blockIdx.x * (blockDim.x >> 5) + (threadIdx.x >> 5);
  const int ntF  = F >> 6;                  // 64-wide column bands
  const int tM   = wave / ntF;
  const int tB   = wave - tM * ntF;
  const int row0 = tM << 4;
  if (row0 >= n) return;                    // wave-uniform: EXEC stays full

  const int hi16 = lane >> 4;
  const int l15  = lane & 15;
  int arow = row0 + l15;
  arow = (arow < n) ? arow : (n - 1);       // clamp: padding rows never stored
  const int kbA = hi16 * 8;
  const float* __restrict__ arp = A + (size_t)arow * K;

  const v16h* __restrict__ BH = (const v16h*)Bh;
  const v16h* __restrict__ BL = (const v16h*)Bl;

  v8f acc0 = {}, acc1 = {}, acc2 = {}, acc3 = {};

#define B_STEP(ACC, GOFF)                                                              \
  {                                                                                    \
    v16h bh = BH[gb + (GOFF)];                                                         \
    v16h bl = BL[gb + (GOFF)];                                                         \
    ACC = __builtin_amdgcn_wmma_f32_16x16x32_f16(false, al, false, bh, (short)0, ACC, false, false); \
    ACC = __builtin_amdgcn_wmma_f32_16x16x32_f16(false, ah, false, bl, (short)0, ACC, false, false); \
    ACC = __builtin_amdgcn_wmma_f32_16x16x32_f16(false, ah, false, bh, (short)0, ACC, false, false); \
  }

  const int kfull = K >> 5;
  for (int kc = 0; kc < kfull; ++kc) {
    const int k0 = kc << 5;
    const float* __restrict__ p = arp + k0 + kbA;
    __builtin_prefetch(p + 32, 0, 1);       // global_prefetch_b8: next chunk
    v16h ah, al;
#pragma unroll
    for (int i = 0; i < 16; ++i) {
      float av = (i < 8) ? p[i] : p[8 + i]; // two contiguous 8-float runs
      _Float16 avh = (_Float16)av;
      ah[i] = avh;
      al[i] = (_Float16)(av - (float)avh);
    }
    const size_t gb = (size_t)(kc * 2 + hi16) * F + (tB << 6) + l15;
    B_STEP(acc0, 0)
    B_STEP(acc1, 16)
    B_STEP(acc2, 32)
    B_STEP(acc3, 48)
  }

  if (K & 31) {                             // tail chunk: branch-free guards
    const int kc = kfull;
    const int k0 = kc << 5;
    v16h ah, al;
#pragma unroll
    for (int i = 0; i < 16; ++i) {
      int ka = k0 + kbA + (i < 8 ? i : i + 8);
      int kcl = (ka < K) ? ka : (K - 1);    // clamped index: load always legal
      float av = arp[kcl];
      av = (ka < K) ? av : 0.0f;            // VALU select, no EXEC masking
      _Float16 avh = (_Float16)av;
      ah[i] = avh;
      al[i] = (_Float16)(av - (float)avh);
    }
    const size_t gb = (size_t)(kc * 2 + hi16) * F + (tB << 6) + l15;
    B_STEP(acc0, 0)
    B_STEP(acc1, 16)
    B_STEP(acc2, 32)
    B_STEP(acc3, 48)
  }
#undef B_STEP

  // C/D layout: VGPR v -> row = row0 + v + 8*hi16, col = subtile*16 + l15
  const int crow0 = row0 + (hi16 << 3);
#define STORE_TILE(ACC, J)                                                   \
  {                                                                          \
    const int col = (tB << 6) + ((J) << 4) + l15;                            \
    const float bv = bias ? bias[col] : 0.0f;                                \
    _Pragma("unroll")                                                        \
    for (int v = 0; v < 8; ++v) {                                            \
      int r = crow0 + v;                                                     \
      if (r < n) {                                                           \
        float val = ACC[v] + bv;                                             \
        if (fuse_relu) val = fmaxf(val, 0.0f);                               \
        C[(size_t)r * F + col] = val;                                        \
      }                                                                      \
    }                                                                        \
  }
  STORE_TILE(acc0, 0)
  STORE_TILE(acc1, 1)
  STORE_TILE(acc2, 2)
  STORE_TILE(acc3, 3)
#undef STORE_TILE
}

// ---------------------------------------------------------------------------
// GCN aggregation: out = dinv_d * sum_edges(dinv_s * h_s) incl. self loop + b
// ---------------------------------------------------------------------------
__global__ void k_selfinit(const float* __restrict__ h, const float* __restrict__ dinv,
                           const float* __restrict__ b, float* __restrict__ out,
                           int n, int F) {
  long long t = (long long)blockIdx.x * blockDim.x + threadIdx.x;
  if (t >= (long long)n * F) return;
  int i = (int)(t / F);
  int f = (int)(t - (long long)i * F);
  float di = dinv[i];
  out[t] = h[t] * di * di + b[f];
}

__global__ void k_edge_agg(const int* __restrict__ src, const int* __restrict__ dst, int e,
                           const float* __restrict__ h, const float* __restrict__ dinv,
                           float* __restrict__ out, int F) {
  long long t = (long long)blockIdx.x * blockDim.x + threadIdx.x;
  if (t >= (long long)e * F) return;
  int ei = (int)(t / F);
  int f  = (int)(t - (long long)ei * F);
  int s = src[ei];
  int d = dst[ei];
  atomicAdd(&out[(long long)d * F + f], h[(long long)s * F + f] * dinv[s] * dinv[d]);
}

// ---------------------------------------------------------------------------
// Batch norm: coalesced partial sums, per-feature atomics, finalize, apply.
// ---------------------------------------------------------------------------
__global__ void k_bn_partial(const float* __restrict__ X, int n, int F,
                             float* __restrict__ sum, float* __restrict__ sumsq) {
  const int f = threadIdx.x % F;            // F divides 256 (128 or 64)
  const int rlane = threadIdx.x / F;
  const int rowsPerBlk = 256 / F;
  long long row = (long long)blockIdx.x * rowsPerBlk + rlane;
  const long long rstride = (long long)gridDim.x * rowsPerBlk;
  float s = 0.0f, s2 = 0.0f;
  for (; row < n; row += rstride) {
    float x = X[row * F + f];
    s += x; s2 += x * x;
  }
  __shared__ float sh[256], sh2[256];
  sh[threadIdx.x] = s; sh2[threadIdx.x] = s2;
  __syncthreads();
  for (int o = 128; o >= F; o >>= 1) {
    if (threadIdx.x < o) {
      sh[threadIdx.x]  += sh[threadIdx.x + o];
      sh2[threadIdx.x] += sh2[threadIdx.x + o];
    }
    __syncthreads();
  }
  if (threadIdx.x < F) {
    atomicAdd(&sum[f],   sh[threadIdx.x]);
    atomicAdd(&sumsq[f], sh2[threadIdx.x]);
  }
}

__global__ void k_bn_finalize(const float* __restrict__ sum, const float* __restrict__ sumsq,
                              int n, int F, float* __restrict__ mean, float* __restrict__ var) {
  int f = blockIdx.x * blockDim.x + threadIdx.x;
  if (f < F) {
    float m = sum[f] / (float)n;
    mean[f] = m;
    var[f]  = sumsq[f] / (float)n - m * m;
  }
}

__global__ void k_bn_apply(float* __restrict__ X, int n, int F,
                           const float* __restrict__ mean, const float* __restrict__ var,
                           const float* __restrict__ g, const float* __restrict__ be,
                           int relu) {
  long long t = (long long)blockIdx.x * blockDim.x + threadIdx.x;
  if (t >= (long long)n * F) return;
  int f = (int)(t % F);
  float y = (X[t] - mean[f]) * rsqrtf(var[f] + EPS_BN) * g[f] + be[f];
  if (relu) y = fmaxf(y, 0.0f);
  X[t] = y;
}

// ---------------------------------------------------------------------------
// Host driver
// ---------------------------------------------------------------------------
static inline int cdiv(long long a, long long b) { return (int)((a + b - 1) / b); }

extern "C" void kernel_launch(void* const* d_in, const int* in_sizes, int n_in,
                              void* d_out, int out_size, void* d_ws, size_t ws_size,
                              hipStream_t stream) {
  const float* x    = (const float*)d_in[0];
  const int*   edge = (const int*)d_in[1];
  const float* W1   = (const float*)d_in[2];
  const float* b1   = (const float*)d_in[3];
  const float* g1   = (const float*)d_in[4];
  const float* be1  = (const float*)d_in[5];
  const float* W2   = (const float*)d_in[6];
  const float* b2   = (const float*)d_in[7];
  const float* g2   = (const float*)d_in[8];
  const float* be2  = (const float*)d_in[9];
  const float* Wp1  = (const float*)d_in[10];
  const float* bp1  = (const float*)d_in[11];
  const float* Wp2  = (const float*)d_in[12];
  const float* bp2  = (const float*)d_in[13];

  const int N = in_sizes[0] / IN_F;
  const int E = in_sizes[1] / 2;
  const int* src = edge;       // edge_index row 0
  const int* dst = edge + E;   // edge_index row 1

  // Workspace layout (float units, 64-float (256B) aligned carve-outs)
  float* ws    = (float*)d_ws;
  float* bufA  = ws;                                 // N*128 (h_pre L1, reused as MLP hidden)
  float* bufB  = bufA + (size_t)N * HID_F;           // N*128 (h1 aggregated / post-BN)
  float* bufC  = bufB + (size_t)N * HID_F;           // N*64  (h_pre L2)
  float* dinv  = bufC + (size_t)N * OUT_F;           // N
  size_t off   = ((size_t)N * (HID_F + HID_F + OUT_F) + (size_t)N + 63) & ~(size_t)63;
  float* sum   = ws + off;                           // 128
  float* sumsq = sum + HID_F;                        // 128
  float* mean  = sumsq + HID_F;                      // 128
  float* var   = mean + HID_F;                       // 128
  // packed-B scratch: max Kpad=192 -> 192*128 halves = 24576; reserve 32768 each
  _Float16* Bh = (_Float16*)(var + HID_F);           // 32768 halves (64KB)
  _Float16* Bl = Bh + 32768;                         // 32768 halves (64KB)

  float* h2 = (float*)d_out;                         // N*64 (output 1)
  float* z  = h2 + (size_t)N * OUT_F;                // N*64 (output 2)

  const int mt = cdiv(N, 16);                        // 16-row bands

  // --- degrees & symmetric norms ---
  k_fill<<<cdiv(N, 256), 256, 0, stream>>>(dinv, N, 1.0f);
  k_deg<<<cdiv(E, 256), 256, 0, stream>>>(dst, E, dinv);
  k_dinv<<<cdiv(N, 256), 256, 0, stream>>>(dinv, N);

  // --- layer 1: h_pre = x @ W1 (WMMA), aggregate, BN+ReLU ---
  {
    const int Kpad = 192;                            // ceil(165/32)*32
    k_pack_w<<<cdiv((long long)Kpad * HID_F, 256), 256, 0, stream>>>(W1, IN_F, HID_F, Kpad, Bh, Bl);
    int waves = mt * (HID_F / 64);
    k_wmma_gemm<<<cdiv(waves, 4), 128, 0, stream>>>(x, Bh, Bl, nullptr, bufA, N, IN_F, HID_F, 0);
  }
  k_selfinit<<<cdiv((long long)N * HID_F, 256), 256, 0, stream>>>(bufA, dinv, b1, bufB, N, HID_F);
  k_edge_agg<<<cdiv((long long)E * HID_F, 256), 256, 0, stream>>>(src, dst, E, bufA, dinv, bufB, HID_F);
  k_fill<<<1, 256, 0, stream>>>(sum, HID_F, 0.0f);
  k_fill<<<1, 256, 0, stream>>>(sumsq, HID_F, 0.0f);
  k_bn_partial<<<1024, 256, 0, stream>>>(bufB, N, HID_F, sum, sumsq);
  k_bn_finalize<<<1, 256, 0, stream>>>(sum, sumsq, N, HID_F, mean, var);
  k_bn_apply<<<cdiv((long long)N * HID_F, 256), 256, 0, stream>>>(bufB, N, HID_F, mean, var, g1, be1, 1);

  // --- layer 2: h_pre = h1 @ W2 (WMMA), aggregate into d_out, BN (no ReLU) ---
  {
    k_pack_w<<<cdiv((long long)HID_F * OUT_F, 256), 256, 0, stream>>>(W2, HID_F, OUT_F, HID_F, Bh, Bl);
    int waves = mt * (OUT_F / 64);
    k_wmma_gemm<<<cdiv(waves, 4), 128, 0, stream>>>(bufB, Bh, Bl, nullptr, bufC, N, HID_F, OUT_F, 0);
  }
  k_selfinit<<<cdiv((long long)N * OUT_F, 256), 256, 0, stream>>>(bufC, dinv, b2, h2, N, OUT_F);
  k_edge_agg<<<cdiv((long long)E * OUT_F, 256), 256, 0, stream>>>(src, dst, E, bufC, dinv, h2, OUT_F);
  k_fill<<<1, 256, 0, stream>>>(sum, OUT_F, 0.0f);
  k_fill<<<1, 256, 0, stream>>>(sumsq, OUT_F, 0.0f);
  k_bn_partial<<<1024, 256, 0, stream>>>(h2, N, OUT_F, sum, sumsq);
  k_bn_finalize<<<1, 256, 0, stream>>>(sum, sumsq, N, OUT_F, mean, var);
  k_bn_apply<<<cdiv((long long)N * OUT_F, 256), 256, 0, stream>>>(h2, N, OUT_F, mean, var, g2, be2, 0);

  // --- MLP head: z = relu(h2 @ Wp1 + bp1) @ Wp2 + bp2 (bias+ReLU fused) ---
  {
    int waves = mt * (OUT_F / 64);
    k_pack_w<<<cdiv((long long)OUT_F * OUT_F, 256), 256, 0, stream>>>(Wp1, OUT_F, OUT_F, OUT_F, Bh, Bl);
    k_wmma_gemm<<<cdiv(waves, 4), 128, 0, stream>>>(h2, Bh, Bl, bp1, bufA, N, OUT_F, OUT_F, 1);
    k_pack_w<<<cdiv((long long)OUT_F * OUT_F, 256), 256, 0, stream>>>(Wp2, OUT_F, OUT_F, OUT_F, Bh, Bl);
    k_wmma_gemm<<<cdiv(waves, 4), 128, 0, stream>>>(bufA, Bh, Bl, bp2, z, N, OUT_F, OUT_F, 0);
  }
}
